// LinearAttentionBlock_249108103320
// MI455X (gfx1250) — matrix operand order repaired and checked by reading, not compiled, not run on hardware
//
#include <hip/hip_runtime.h>
#include <hip/hip_bf16.h>
#include <cmath>

typedef _Float16 half8  __attribute__((ext_vector_type(8)));
typedef _Float16 half16 __attribute__((ext_vector_type(16)));
typedef float    float8 __attribute__((ext_vector_type(8)));

#define DIMX     384
#define HEADS    6
#define HEAD_DIM 64
#define HIDDEN   1536
#define NTOK     4096
#define BATCH    8
#define ROWS     (BATCH * NTOK)       /* 32768 tokens */
#define BHTOT    (BATCH * HEADS)      /* 48 (b,h) pairs */

__device__ __forceinline__ float warp_allreduce(float v) {
#pragma unroll
  for (int m = 16; m; m >>= 1) v += __shfl_xor(v, m, 32);
  return v;
}

// ---------------------------------------------------------------------------
// Pack f32 row-major weight [K,N] into WMMA-B fragment layout (f16):
// packed[((kt*nTiles + nt)*32 + lane)*16 + e] = W[kt*32 + (lane&16?16:0) + e][nt*16 + (lane&15)]
// => each lane's whole 16-half fragment is contiguous (one 32B load).
// ---------------------------------------------------------------------------
__global__ void pack_w(const float* __restrict__ W, _Float16* __restrict__ P,
                       int K, int N) {
  int total = K * N;
  int p = blockIdx.x * blockDim.x + threadIdx.x;
  if (p >= total) return;
  int e    = p & 15;
  int lane = (p >> 4) & 31;
  int tf   = p >> 9;
  int ntiles = N >> 4;
  int nt = tf % ntiles;
  int kt = tf / ntiles;
  int k = kt * 32 + ((lane & 16) ? 16 : 0) + e;
  int n = nt * 16 + (lane & 15);
  P[p] = (_Float16)W[(size_t)k * N + n];
}

// ---------------------------------------------------------------------------
// LayerNorm: one wave per 384-wide row, f32 in -> f16 out.
// ---------------------------------------------------------------------------
__global__ __launch_bounds__(256) void ln_rows(const float* __restrict__ x,
                                               const float* __restrict__ g,
                                               const float* __restrict__ b,
                                               _Float16* __restrict__ out,
                                               int rows) {
  int wave = threadIdx.x >> 5;
  int lane = threadIdx.x & 31;
  int row  = blockIdx.x * 8 + wave;
  if (row >= rows) return;
  const float* xr = x + (size_t)row * DIMX;
  float v[12];
  float s = 0.f, sq = 0.f;
#pragma unroll
  for (int j = 0; j < 12; ++j) {
    v[j] = xr[lane + j * 32];
    s += v[j];
    sq += v[j] * v[j];
  }
  s  = warp_allreduce(s);
  sq = warp_allreduce(sq);
  float mean = s * (1.0f / DIMX);
  float var  = sq * (1.0f / DIMX) - mean * mean;
  float rstd = rsqrtf(var + 1e-5f);
  _Float16* o = out + (size_t)row * DIMX;
#pragma unroll
  for (int j = 0; j < 12; ++j) {
    int i = lane + j * 32;
    o[i] = (_Float16)((v[j] - mean) * rstd * g[i] + b[i]);
  }
}

// ---------------------------------------------------------------------------
// WMMA GEMM: C[M,N] = A[M,K](f16,row-major) * Bpacked[K,N](f16, fragment
// layout) with fused epilogues. Block = 256 threads = 8 waves (4x2 wave grid),
// wave tile = 32x32 (2x2 v_wmma_f32_16x16x32_f16), block tile = 128x64.
// K loop is 2-stage software pipelined (K/32 must be even: 12, 2, 48 here).
//
// EPI 0: qkv split  : +bias, phi=elu+1 on q/k, scatter to [B,H,N,64] f16 bufs
// EPI 1: attention  : *z[row], scatter heads back into [B,N,384] f16
// EPI 2: residual   : +bias, fout = resid + val (f32)
// EPI 3: gelu       : +bias, h0 = gelu(val) (f16)
// ---------------------------------------------------------------------------
template <int EPI, bool HAS_BIAS>
__global__ __launch_bounds__(256) void wmma_gemm(
    const _Float16* __restrict__ A, const _Float16* __restrict__ Bp,
    const float* __restrict__ bias, int M, int N, int K, int tokPerBH,
    int bPerBH, const float* __restrict__ zvec, const float* __restrict__ resid,
    float* __restrict__ fout, _Float16* __restrict__ h0,
    _Float16* __restrict__ h1, _Float16* __restrict__ h2) {
  const int lane = threadIdx.x & 31;
  const int wave = threadIdx.x >> 5;
  const int wm = wave & 3;
  const int wn = wave >> 2;
  const int rowBase = blockIdx.x * 128 + wm * 32;
  const int colBase = blockIdx.y * 64 + wn * 32;
  const int nTiles = N >> 4;

  const _Float16* Bq = Bp;
  if (tokPerBH > 0) {  // per-(b,h) B matrix (attention kv-state)
    int bh = (blockIdx.x * 128) / tokPerBH;
    Bq += (size_t)bh * bPerBH;
  }

  const int klo  = (lane < 16) ? 0 : 16;  // contiguous A-layout k offset (matches packed B)
  const int mrow = lane & 15;
  const _Float16* aBase0 = A + (size_t)(rowBase + mrow) * K + klo;
  const _Float16* aBase1 = A + (size_t)(rowBase + 16 + mrow) * K + klo;
  const _Float16* bBase  = Bq + ((size_t)((colBase >> 4)) * 32 + lane) * 16;

  auto loadFrags = [&](int kb, half16 (&af)[2], half16 (&bf)[2]) {
    const int k0 = kb * 32;
#pragma unroll
    for (int mt = 0; mt < 2; ++mt) {
      const _Float16* ap = (mt ? aBase1 : aBase0) + k0;
      half8 lo = *(const half8*)ap;
      half8 hi = *(const half8*)(ap + 8);
      af[mt] = __builtin_shufflevector(lo, hi, 0, 1, 2, 3, 4, 5, 6, 7, 8, 9,
                                       10, 11, 12, 13, 14, 15);
      __builtin_prefetch((const void*)(ap + 64), 0, 1);  // two k-tiles ahead
    }
#pragma unroll
    for (int nt = 0; nt < 2; ++nt)
      bf[nt] = *(const half16*)(bBase + ((size_t)kb * nTiles + nt) * 512);
  };

  const float8 zero8 = {0.f, 0.f, 0.f, 0.f, 0.f, 0.f, 0.f, 0.f};
  float8 acc[2][2];
#pragma unroll
  for (int i = 0; i < 2; ++i)
#pragma unroll
    for (int j = 0; j < 2; ++j) acc[i][j] = zero8;

  auto mmaStep = [&](half16 (&af)[2], half16 (&bf)[2]) {
#pragma unroll
    for (int mt = 0; mt < 2; ++mt)
#pragma unroll
      for (int nt = 0; nt < 2; ++nt)
        acc[mt][nt] = __builtin_amdgcn_wmma_f32_16x16x32_f16(
            false, af[mt], false, bf[nt], (short)0, acc[mt][nt], false, false);
  };

  // --- 2-stage software-pipelined K loop (kIters is even by construction) ---
  const int kIters = K >> 5;
  half16 a0[2], b0[2], a1[2], b1[2];
  loadFrags(0, a0, b0);
  for (int kb = 0; kb + 2 <= kIters; kb += 2) {
    loadFrags(kb + 1, a1, b1);
    mmaStep(a0, b0);
    if (kb + 2 < kIters) loadFrags(kb + 2, a0, b0);
    mmaStep(a1, b1);
  }

  // Epilogue: 32-bit C/D layout — lane<16: rows r, N=lane; lane>=16: rows r+8.
#pragma unroll
  for (int mt = 0; mt < 2; ++mt) {
#pragma unroll
    for (int nt = 0; nt < 2; ++nt) {
#pragma unroll
      for (int r = 0; r < 8; ++r) {
        int row = rowBase + mt * 16 + ((lane < 16) ? r : r + 8);
        int col = colBase + nt * 16 + (lane & 15);
        float val = acc[mt][nt][r];
        if constexpr (HAS_BIAS) val += bias[col];
        if constexpr (EPI == 0) {
          int bi = row >> 12;  // / NTOK
          int ntk = row & (NTOK - 1);
          int which = col / DIMX;
          int rem = col - which * DIMX;
          int hh = rem >> 6;
          int dd = rem & 63;
          size_t dst = ((size_t)(bi * HEADS + hh) * NTOK + ntk) * HEAD_DIM + dd;
          if (which == 0)
            h0[dst] = (_Float16)(val > 0.f ? val + 1.f : __expf(val));
          else if (which == 1)
            h1[dst] = (_Float16)(val > 0.f ? val + 1.f : __expf(val));
          else
            h2[dst] = (_Float16)val;
        } else if constexpr (EPI == 1) {
          val *= zvec[row];
          int bh = row >> 12;
          int ntk = row & (NTOK - 1);
          int bi = bh / HEADS;
          int hh = bh - bi * HEADS;
          size_t dst = ((size_t)(bi * NTOK + ntk)) * DIMX + hh * HEAD_DIM + col;
          h0[dst] = (_Float16)val;
        } else if constexpr (EPI == 2) {
          size_t idx = (size_t)row * N + col;
          fout[idx] = resid[idx] + val;
        } else {
          float xx = val;
          float t = 0.7978845608028654f * (xx + 0.044715f * xx * xx * xx);
          float ge = 0.5f * xx * (1.f + tanhf(t));
          h0[(size_t)row * N + col] = (_Float16)ge;
        }
      }
    }
  }
}

// ---------------------------------------------------------------------------
// Per-(b,h): kv[64][64] = sum_n phiK[n] (outer) V[n]; ksum[64] = sum_n phiK[n].
// kv is written directly in packed WMMA-B layout for the attention GEMM.
// ---------------------------------------------------------------------------
__global__ __launch_bounds__(256) void kv_reduce(const _Float16* __restrict__ phiK,
                                                 const _Float16* __restrict__ V,
                                                 _Float16* __restrict__ kvpack,
                                                 float* __restrict__ ksum) {
  __shared__ _Float16 lk[16 * 64];
  __shared__ _Float16 lv[16 * 64];
  int bh = blockIdx.x;
  int t = threadIdx.x;
  int dv = t & 63;
  int dkb = t >> 6;  // 0..3 : owns dk = dkb*16 .. dkb*16+15
  float acc[16];
#pragma unroll
  for (int i = 0; i < 16; ++i) acc[i] = 0.f;
  float ks = 0.f;
  const _Float16* kb = phiK + (size_t)bh * NTOK * HEAD_DIM;
  const _Float16* vb = V + (size_t)bh * NTOK * HEAD_DIM;
  for (int n0 = 0; n0 < NTOK; n0 += 16) {
#pragma unroll
    for (int j = 0; j < 4; ++j) {
      int i = t + j * 256;
      lk[i] = kb[n0 * 64 + i];
      lv[i] = vb[n0 * 64 + i];
    }
    __syncthreads();
#pragma unroll 4
    for (int r = 0; r < 16; ++r) {
      float vv = (float)lv[r * 64 + dv];
#pragma unroll
      for (int i = 0; i < 16; ++i)
        acc[i] += (float)lk[r * 64 + dkb * 16 + i] * vv;
    }
    if (t < 64) {
#pragma unroll 4
      for (int r = 0; r < 16; ++r) ks += (float)lk[r * 64 + t];
    }
    __syncthreads();
  }
#pragma unroll
  for (int i = 0; i < 16; ++i) {
    int dk = dkb * 16 + i;
    int kt = dk >> 5;
    int kk = dk & 31;
    int nt = dv >> 4;
    int nn = dv & 15;
    int lanep = ((kk >> 4) << 4) + nn;
    int e = kk & 15;
    size_t p = (size_t)bh * 4096 + (((size_t)(kt * 4 + nt) * 32 + lanep) << 4) + e;
    kvpack[p] = (_Float16)acc[i];
  }
  if (t < 64) ksum[bh * 64 + t] = ks;
}

// ---------------------------------------------------------------------------
// z[rho] = 1 / (phiQ[rho] . ksum[bh] + 1e-6); one wave per token.
// ---------------------------------------------------------------------------
__global__ __launch_bounds__(256) void z_kernel(const _Float16* __restrict__ phiQ,
                                                const float* __restrict__ ksum,
                                                float* __restrict__ z,
                                                int total) {
  int wave = threadIdx.x >> 5;
  int lane = threadIdx.x & 31;
  int rho = blockIdx.x * 8 + wave;
  if (rho >= total) return;
  int bh = rho >> 12;
  const _Float16* q = phiQ + (size_t)rho * 64;
  const float* ks = ksum + bh * 64;
  float s = (float)q[lane] * ks[lane] + (float)q[lane + 32] * ks[lane + 32];
#pragma unroll
  for (int m = 16; m; m >>= 1) s += __shfl_xor(s, m, 32);
  if (lane == 0) z[rho] = 1.0f / (s + 1e-6f);
}

// ---------------------------------------------------------------------------
extern "C" void kernel_launch(void* const* d_in, const int* in_sizes, int n_in,
                              void* d_out, int out_size, void* d_ws,
                              size_t ws_size, hipStream_t stream) {
  (void)in_sizes; (void)n_in; (void)out_size; (void)ws_size;
  const float* x      = (const float*)d_in[0];
  const float* n1g    = (const float*)d_in[1];
  const float* n1b    = (const float*)d_in[2];
  const float* qkv_w  = (const float*)d_in[3];
  const float* qkv_b  = (const float*)d_in[4];
  const float* proj_w = (const float*)d_in[5];
  const float* proj_b = (const float*)d_in[6];
  const float* n2g    = (const float*)d_in[7];
  const float* n2b    = (const float*)d_in[8];
  const float* fc1_w  = (const float*)d_in[9];
  const float* fc1_b  = (const float*)d_in[10];
  const float* fc2_w  = (const float*)d_in[11];
  const float* fc2_b  = (const float*)d_in[12];
  float* out = (float*)d_out;

  char* ws = (char*)d_ws;
  size_t off = 0;
  auto alloc = [&](size_t bytes) -> char* {
    char* p = ws + off;
    off += (bytes + 255) & ~(size_t)255;
    return p;
  };
  _Float16* hbuf  = (_Float16*)alloc((size_t)ROWS * DIMX * 2);     // LN1 out
  _Float16* phiQ  = (_Float16*)alloc((size_t)ROWS * DIMX * 2);     // [B,H,N,64]
  _Float16* phiK  = (_Float16*)alloc((size_t)ROWS * DIMX * 2);
  _Float16* vbuf  = (_Float16*)alloc((size_t)ROWS * DIMX * 2);
  _Float16* attn  = (_Float16*)alloc((size_t)ROWS * DIMX * 2);     // [B,N,384]
  _Float16* h2    = (_Float16*)alloc((size_t)ROWS * DIMX * 2);     // LN2 out
  _Float16* h3    = (_Float16*)alloc((size_t)ROWS * HIDDEN * 2);   // gelu out
  float*    x2    = (float*)alloc((size_t)ROWS * DIMX * 4);        // mid resid
  float*    zbuf  = (float*)alloc((size_t)BHTOT * NTOK * 4);
  float*    ksum  = (float*)alloc((size_t)BHTOT * 64 * 4);
  _Float16* kvpk  = (_Float16*)alloc((size_t)BHTOT * 4096 * 2);    // packed kv
  _Float16* pQKV  = (_Float16*)alloc((size_t)DIMX * 3 * DIMX * 2);
  _Float16* pPROJ = (_Float16*)alloc((size_t)DIMX * DIMX * 2);
  _Float16* pFC1  = (_Float16*)alloc((size_t)DIMX * HIDDEN * 2);
  _Float16* pFC2  = (_Float16*)alloc((size_t)HIDDEN * DIMX * 2);

  dim3 blk(256);
  // --- weight packing (f32 -> WMMA-B f16 fragment layout) ---
  pack_w<<<(DIMX * 3 * DIMX + 255) / 256, blk, 0, stream>>>(qkv_w, pQKV, DIMX, 3 * DIMX);
  pack_w<<<(DIMX * DIMX + 255) / 256, blk, 0, stream>>>(proj_w, pPROJ, DIMX, DIMX);
  pack_w<<<(DIMX * HIDDEN + 255) / 256, blk, 0, stream>>>(fc1_w, pFC1, DIMX, HIDDEN);
  pack_w<<<(HIDDEN * DIMX + 255) / 256, blk, 0, stream>>>(fc2_w, pFC2, HIDDEN, DIMX);

  // --- LN1 ---
  ln_rows<<<ROWS / 8, blk, 0, stream>>>(x, n1g, n1b, hbuf, ROWS);

  // --- QKV GEMM + phi split ---
  wmma_gemm<0, true><<<dim3(ROWS / 128, (3 * DIMX) / 64), blk, 0, stream>>>(
      hbuf, pQKV, qkv_b, ROWS, 3 * DIMX, DIMX, 0, 0, nullptr, nullptr, nullptr,
      phiQ, phiK, vbuf);

  // --- kv state + ksum per (b,h) ---
  kv_reduce<<<BHTOT, blk, 0, stream>>>(phiK, vbuf, kvpk, ksum);

  // --- z = 1/(q.ksum + eps) ---
  z_kernel<<<(BHTOT * NTOK) / 8, blk, 0, stream>>>(phiQ, ksum, zbuf, BHTOT * NTOK);

  // --- attn = z * (phiQ @ kv), re-interleave heads ---
  wmma_gemm<1, false><<<dim3((BHTOT * NTOK) / 128, 1), blk, 0, stream>>>(
      phiQ, kvpk, nullptr, BHTOT * NTOK, HEAD_DIM, HEAD_DIM, NTOK, 4096, zbuf,
      nullptr, nullptr, attn, nullptr, nullptr);

  // --- proj GEMM + residual -> x2 ---
  wmma_gemm<2, true><<<dim3(ROWS / 128, DIMX / 64), blk, 0, stream>>>(
      attn, pPROJ, proj_b, ROWS, DIMX, DIMX, 0, 0, nullptr, x, x2, nullptr,
      nullptr, nullptr);

  // --- LN2 ---
  ln_rows<<<ROWS / 8, blk, 0, stream>>>(x2, n2g, n2b, h2, ROWS);

  // --- FC1 + GELU ---
  wmma_gemm<3, true><<<dim3(ROWS / 128, HIDDEN / 64), blk, 0, stream>>>(
      h2, pFC1, fc1_b, ROWS, HIDDEN, DIMX, 0, 0, nullptr, nullptr, nullptr, h3,
      nullptr, nullptr);

  // --- FC2 + residual -> out ---
  wmma_gemm<2, true><<<dim3(ROWS / 128, DIMX / 64), blk, 0, stream>>>(
      h3, pFC2, fc2_b, ROWS, DIMX, HIDDEN, 0, 0, nullptr, x2, out, nullptr,
      nullptr, nullptr);
}
